// PSP_HDC_DUC_76613626626140
// MI455X (gfx1250) — matrix operand order repaired and checked by reading, not compiled, not run on hardware
//
#include <hip/hip_runtime.h>
#include <hip/hip_bf16.h>

// ---------------------------------------------------------------------------
// PSP/HDC/DUC pipeline for MI455X (gfx1250, wave32, WMMA).
// conv1 (512x5632 GEMM, 94.5 GFLOP) -> BN+ReLU -> conv2 (384x512 GEMM,
// shifted window) -> +bias -> DUC pixel shuffle -> softmax(6) -> bilinear.
// Big GEMM in bf16 WMMA (v_wmma_f32_16x16x32_bf16), fp32 accumulation.
// Intermediate h is stored n-major ([b][n][m], bf16) so that (a) the GEMM1
// epilogue writes one packed global_store_b128 per accumulator tile and
// (b) GEMM2's B tiles are k-contiguous (vector loads, no transpose).
// ---------------------------------------------------------------------------

typedef __attribute__((ext_vector_type(16))) __bf16 v16bf;
typedef __attribute__((ext_vector_type(8)))  __bf16 v8bf;
typedef __attribute__((ext_vector_type(4)))  __bf16 v4bf;
typedef __attribute__((ext_vector_type(8)))  float  v8f;

#define K1     5632
#define M1     512
#define M2     384
#define NSPAT  1024        // 32*32 spatial positions per batch
#define NBATCH 16

// ---- WMMA fragment loaders (ISA 7.12.2 bf16 layouts) ----------------------
// A 16x32 bf16: lanes 0-15 row=lane hold K {0..7, 16..23}; lanes 16-31 hold
// K {8..15, 24..31}.  As is row-major [m][k], 32 bf16 per row.
__device__ __forceinline__ v16bf load_a_frag(const __bf16* __restrict__ As,
                                             int mt, int lane) {
  const int row = mt + (lane & 15);
  const int kh  = lane >> 4;                  // 0 or 1
  const v8bf lo = *(const v8bf*)(As + row * 32 + kh * 8);
  const v8bf hi = *(const v8bf*)(As + row * 32 + 16 + kh * 8);
  v16bf a;
#pragma unroll
  for (int i = 0; i < 8; ++i) { a[i] = lo[i]; a[i + 8] = hi[i]; }
  return a;
}

// B 32x16 bf16: lane n = lane&15; lanes 0-15 hold K {0..7, 16..23}, lanes 16-31
// hold K {8..15, 24..31} (same order as A).  Bs is [n][k] with row stride 40 bf16 (80 B:
// 16-B aligned, spreads LDS banks).
__device__ __forceinline__ v16bf load_b_frag(const __bf16* __restrict__ Bs,
                                             int nt, int lane) {
  const int n  = nt + (lane & 15);
  const int kh = lane >> 4;
  const v8bf lo = *(const v8bf*)(Bs + n * 40 + kh * 8);
  const v8bf hi = *(const v8bf*)(Bs + n * 40 + 16 + kh * 8);
  v16bf b;
#pragma unroll
  for (int i = 0; i < 8; ++i) { b[i] = lo[i]; b[i + 8] = hi[i]; }
  return b;
}

// ---- kernel 0: fold BN into scale/shift, border constant ------------------
__global__ void prep_kernel(const float* __restrict__ gamma,
                            const float* __restrict__ beta,
                            const float* __restrict__ mean,
                            const float* __restrict__ var,
                            float* __restrict__ scale,
                            float* __restrict__ shift,
                            __bf16* __restrict__ hpad) {
  const int m = blockIdx.x * blockDim.x + threadIdx.x;
  if (m < M1) {
    const float s  = gamma[m] * rsqrtf(var[m] + 1e-5f);
    const float sh = beta[m] - mean[m] * s;
    scale[m] = s;
    shift[m] = sh;
    hpad[m]  = (__bf16)fmaxf(sh, 0.0f);   // conv1 output at padded border
  }
}

// ---- kernel 1: conv1 GEMM (M=512, K=5632, N=1024/batch) + BN + ReLU -------
__global__ __launch_bounds__(256, 2)
void gemm1_bn_relu(const float* __restrict__ x, const float* __restrict__ w1,
                   const float* __restrict__ scale,
                   const float* __restrict__ shift,
                   __bf16* __restrict__ h) {
  const int tileN = blockIdx.x << 7;          // 0..896
  const int tileM = blockIdx.y << 7;          // 0..384
  const int b     = blockIdx.z;
  const float* __restrict__ xb = x + (size_t)b * K1 * NSPAT;

  __shared__ __align__(16) __bf16 As[2][128 * 32];   // 8 KB each
  __shared__ __align__(16) __bf16 Bs[2][128 * 40];   // 10 KB each

  const int t    = threadIdx.x;
  const int lane = t & 31;
  const int wave = t >> 5;                    // 8 waves
  const int wm   = wave >> 2;                 // 0..1  -> 64-row strip
  const int wn   = wave & 3;                  // 0..3  -> 32-col strip

  const int ar   = t >> 1;                    // A row 0..127
  const int ac   = (t & 1) << 4;              // A col chunk 0/16
  const int kgrp = (t >> 5) << 2;             // B k rows: kgrp..kgrp+3
  const int ngrp = (t & 31) << 2;             // B n cols: ngrp..ngrp+3

  v8f acc[4][2];
#pragma unroll
  for (int i = 0; i < 4; ++i)
#pragma unroll
    for (int j = 0; j < 2; ++j)
#pragma unroll
      for (int r = 0; r < 8; ++r) acc[i][j][r] = 0.0f;

  const float* pA = w1 + (size_t)(tileM + ar) * K1 + ac;
  const float* pB = xb + (size_t)kgrp * NSPAT + tileN + ngrp;

  float4 ga[4], gb[4];
  auto fetch = [&](int kq) {
    const float4* gAp = (const float4*)(pA + kq);
    ga[0] = gAp[0]; ga[1] = gAp[1]; ga[2] = gAp[2]; ga[3] = gAp[3];
    // 4(k) x 4(n) micro-block of x
#pragma unroll
    for (int i = 0; i < 4; ++i)
      gb[i] = *(const float4*)(pB + (size_t)(kq + i) * NSPAT);
    __builtin_prefetch(pA + kq + 32, 0, 0);                  // global_prefetch
    __builtin_prefetch(pB + (size_t)(kq + 32) * NSPAT, 0, 0);
  };
  auto commit = [&](int buf) {
    // A: row-major, 16 bf16 per thread, two b128 stores
    v8bf lo, hi;
    lo[0] = (__bf16)ga[0].x; lo[1] = (__bf16)ga[0].y;
    lo[2] = (__bf16)ga[0].z; lo[3] = (__bf16)ga[0].w;
    lo[4] = (__bf16)ga[1].x; lo[5] = (__bf16)ga[1].y;
    lo[6] = (__bf16)ga[1].z; lo[7] = (__bf16)ga[1].w;
    hi[0] = (__bf16)ga[2].x; hi[1] = (__bf16)ga[2].y;
    hi[2] = (__bf16)ga[2].z; hi[3] = (__bf16)ga[2].w;
    hi[4] = (__bf16)ga[3].x; hi[5] = (__bf16)ga[3].y;
    hi[6] = (__bf16)ga[3].z; hi[7] = (__bf16)ga[3].w;
    __bf16* da = &As[buf][ar * 32 + ac];
    *(v8bf*)da = lo;
    *(v8bf*)(da + 8) = hi;
    // B: register transpose of the 4x4 block -> four b64 stores into [n][k]
    const float fb[4][4] = {{gb[0].x, gb[0].y, gb[0].z, gb[0].w},
                            {gb[1].x, gb[1].y, gb[1].z, gb[1].w},
                            {gb[2].x, gb[2].y, gb[2].z, gb[2].w},
                            {gb[3].x, gb[3].y, gb[3].z, gb[3].w}};
#pragma unroll
    for (int j = 0; j < 4; ++j) {              // n within micro-block
      v4bf col;
#pragma unroll
      for (int i = 0; i < 4; ++i) col[i] = (__bf16)fb[i][j];   // k 0..3
      *(v4bf*)&Bs[buf][(ngrp + j) * 40 + kgrp] = col;
    }
  };

  fetch(0);
  commit(0);
  __syncthreads();

  const int KT = K1 / 32;                      // 176 k-steps
  for (int kk = 0; kk < KT; ++kk) {
    const int cur = kk & 1;
    if (kk + 1 < KT) fetch((kk + 1) * 32);     // global loads one stage ahead

    const __bf16* Ac = As[cur];
    const __bf16* Bc = Bs[cur];
    v16bf af[4], bfr[2];
#pragma unroll
    for (int i = 0; i < 4; ++i) af[i] = load_a_frag(Ac, (wm << 6) + (i << 4), lane);
#pragma unroll
    for (int j = 0; j < 2; ++j) bfr[j] = load_b_frag(Bc, (wn << 5) + (j << 4), lane);
#pragma unroll
    for (int i = 0; i < 4; ++i)
#pragma unroll
      for (int j = 0; j < 2; ++j)
        acc[i][j] = __builtin_amdgcn_wmma_f32_16x16x32_bf16(
            false, af[i], false, bfr[j], (short)0, acc[i][j], false, false);

    if (kk + 1 < KT) commit(cur ^ 1);
    __syncthreads();
  }

  // fused BN + ReLU epilogue; h is n-major [b][n][m] so each accumulator
  // tile packs 8 consecutive m into one global_store_b128.
  const int col  = lane & 15;
  const int rsel = lane >> 4;
#pragma unroll
  for (int i = 0; i < 4; ++i) {
    const int m0 = tileM + (wm << 6) + (i << 4) + (rsel << 3);
#pragma unroll
    for (int j = 0; j < 2; ++j) {
      const int nb = tileN + (wn << 5) + (j << 4) + col;
      v8bf hv;
#pragma unroll
      for (int r = 0; r < 8; ++r) {
        const int m = m0 + r;
        float v = fmaf(acc[i][j][r], scale[m], shift[m]);
        hv[r] = (__bf16)fmaxf(v, 0.0f);
      }
      *(v8bf*)(h + ((size_t)b * NSPAT + nb) * M1 + m0) = hv;
    }
  }
}

// ---- kernel 2: conv2 GEMM (M=384, K=512) over shifted window + bias + DUC -
__global__ __launch_bounds__(256, 2)
void gemm2_bias_duc(const __bf16* __restrict__ h, const float* __restrict__ w2,
                    const float* __restrict__ b2,
                    const __bf16* __restrict__ hpad,
                    float* __restrict__ duc) {
  const int tileN = blockIdx.x << 7;
  const int tileM = blockIdx.y << 7;          // 0,128,256 (M2=384)
  const int b     = blockIdx.z;
  const __bf16* __restrict__ hb = h + (size_t)b * NSPAT * M1;   // [n][m=k]

  __shared__ __align__(16) __bf16 As[2][128 * 32];
  __shared__ __align__(16) __bf16 Bs[2][128 * 40];

  const int t    = threadIdx.x;
  const int lane = t & 31;
  const int wave = t >> 5;
  const int wm   = wave >> 2;
  const int wn   = wave & 3;
  const int ar   = t >> 1;
  const int ac   = (t & 1) << 4;
  const int nl   = t >> 1;                    // B: local n 0..127
  const int kc   = (t & 1) << 4;              // B: k chunk 0/16

  v8f acc[4][2];
#pragma unroll
  for (int i = 0; i < 4; ++i)
#pragma unroll
    for (int j = 0; j < 2; ++j)
#pragma unroll
      for (int r = 0; r < 8; ++r) acc[i][j][r] = 0.0f;

  const float* pA = w2 + (size_t)(tileM + ar) * M1 + ac;
  // shifted-window source pointer for this thread's n (k-contiguous in h):
  // output grid pos (ii,jj) in [:32,:32] of the padded 34x34 conv1 grid;
  // ii==0 || jj==0 is the zero-padded border -> constant hpad[k].
  const int ng = tileN + nl;
  const int ii = ng >> 5, jj = ng & 31;
  const bool border = (ii == 0) || (jj == 0);
  const __bf16* pB =
      (border ? hpad : hb + (size_t)(((ii - 1) << 5) + (jj - 1)) * M1) + kc;

  float4 ga[4];
  v8bf   gb0, gb1;
  auto fetch = [&](int kq) {
    const float4* gAp = (const float4*)(pA + kq);
    ga[0] = gAp[0]; ga[1] = gAp[1]; ga[2] = gAp[2]; ga[3] = gAp[3];
    gb0 = *(const v8bf*)(pB + kq);
    gb1 = *(const v8bf*)(pB + kq + 8);
  };
  auto commit = [&](int buf) {
    v8bf lo, hi;
    lo[0] = (__bf16)ga[0].x; lo[1] = (__bf16)ga[0].y;
    lo[2] = (__bf16)ga[0].z; lo[3] = (__bf16)ga[0].w;
    lo[4] = (__bf16)ga[1].x; lo[5] = (__bf16)ga[1].y;
    lo[6] = (__bf16)ga[1].z; lo[7] = (__bf16)ga[1].w;
    hi[0] = (__bf16)ga[2].x; hi[1] = (__bf16)ga[2].y;
    hi[2] = (__bf16)ga[2].z; hi[3] = (__bf16)ga[2].w;
    hi[4] = (__bf16)ga[3].x; hi[5] = (__bf16)ga[3].y;
    hi[6] = (__bf16)ga[3].z; hi[7] = (__bf16)ga[3].w;
    __bf16* da = &As[buf][ar * 32 + ac];
    *(v8bf*)da = lo;
    *(v8bf*)(da + 8) = hi;
    __bf16* db = &Bs[buf][nl * 40 + kc];
    *(v8bf*)db = gb0;
    *(v8bf*)(db + 8) = gb1;
  };

  fetch(0);
  commit(0);
  __syncthreads();

  const int KT = M1 / 32;                      // 16 k-steps
  for (int kk = 0; kk < KT; ++kk) {
    const int cur = kk & 1;
    if (kk + 1 < KT) fetch((kk + 1) * 32);
    const __bf16* Ac = As[cur];
    const __bf16* Bc = Bs[cur];
    v16bf af[4], bfr[2];
#pragma unroll
    for (int i = 0; i < 4; ++i) af[i] = load_a_frag(Ac, (wm << 6) + (i << 4), lane);
#pragma unroll
    for (int j = 0; j < 2; ++j) bfr[j] = load_b_frag(Bc, (wn << 5) + (j << 4), lane);
#pragma unroll
    for (int i = 0; i < 4; ++i)
#pragma unroll
      for (int j = 0; j < 2; ++j)
        acc[i][j] = __builtin_amdgcn_wmma_f32_16x16x32_bf16(
            false, af[i], false, bfr[j], (short)0, acc[i][j], false, false);
    if (kk + 1 < KT) commit(cur ^ 1);
    __syncthreads();
  }

  // + bias, scatter into DUC pixel-shuffle layout (fp32 logits, 256x256)
  const int col  = lane & 15;
  const int rsel = lane >> 4;
#pragma unroll
  for (int i = 0; i < 4; ++i) {
    const int mb = tileM + (wm << 6) + (i << 4) + (rsel << 3);
#pragma unroll
    for (int j = 0; j < 2; ++j) {
      const int nb = tileN + (wn << 5) + (j << 4) + col;
      const int fy = nb >> 5, fx = nb & 31;
#pragma unroll
      for (int r = 0; r < 8; ++r) {
        const int m = mb + r;
        const float v = acc[i][j][r] + b2[m];
        const int cls = m >> 6;
        const int ry  = (m >> 3) & 7;
        const int rx  = m & 7;
        duc[(((size_t)b * 6 + cls) << 16) + (((fy << 3) + ry) << 8) +
            (fx << 3) + rx] = v;
      }
    }
  }
}

// ---- kernel 3: 6-class softmax at 4 corners + align-corners bilinear ------
__device__ __forceinline__ void corner_probs(const float* __restrict__ duc,
                                             int base, int y, int xx,
                                             float p[6]) {
  if (y < 256 && xx < 256) {
    float l[6];
    float mx = -3.0e38f;
#pragma unroll
    for (int c = 0; c < 6; ++c) {
      l[c] = duc[base + (c << 16) + (y << 8) + xx];
      mx = fmaxf(mx, l[c]);
    }
    float s = 0.0f;
#pragma unroll
    for (int c = 0; c < 6; ++c) { p[c] = __expf(l[c] - mx); s += p[c]; }
    const float inv = 1.0f / s;
#pragma unroll
    for (int c = 0; c < 6; ++c) p[c] *= inv;
  } else {
    // zero-padded DUC region: softmax(0,...,0) = 1/6
#pragma unroll
    for (int c = 0; c < 6; ++c) p[c] = 0.16666667f;
  }
}

__global__ __launch_bounds__(256)
void softmax_bilinear(const float* __restrict__ duc, float* __restrict__ out) {
  const int tid = blockIdx.x * 256 + threadIdx.x;
  const int b   = tid >> 16;
  const int oy  = (tid >> 8) & 255;
  const int ox  = tid & 255;
  const float sc = 303.0f / 255.0f;            // (H-1)/(OH-1), align_corners
  const float fy = oy * sc, fx = ox * sc;
  const int y0 = (int)fy, x0 = (int)fx;
  const float wy = fy - (float)y0, wx = fx - (float)x0;
  const int y1 = min(y0 + 1, 303), x1 = min(x0 + 1, 303);
  const int base = b * 6 * 65536;

  float p00[6], p01[6], p10[6], p11[6];
  corner_probs(duc, base, y0, x0, p00);
  corner_probs(duc, base, y0, x1, p01);
  corner_probs(duc, base, y1, x0, p10);
  corner_probs(duc, base, y1, x1, p11);
#pragma unroll
  for (int c = 0; c < 6; ++c) {
    const float top = p00[c] + (p01[c] - p00[c]) * wx;
    const float bot = p10[c] + (p11[c] - p10[c]) * wx;
    out[base + (c << 16) + (oy << 8) + ox] = top + (bot - top) * wy;
  }
}

// ---------------------------------------------------------------------------
extern "C" void kernel_launch(void* const* d_in, const int* in_sizes, int n_in,
                              void* d_out, int out_size, void* d_ws,
                              size_t ws_size, hipStream_t stream) {
  const float* x     = (const float*)d_in[0];
  const float* w1    = (const float*)d_in[1];
  const float* gamma = (const float*)d_in[2];
  const float* beta  = (const float*)d_in[3];
  const float* mean  = (const float*)d_in[4];
  const float* var   = (const float*)d_in[5];
  const float* w2    = (const float*)d_in[6];
  const float* b2    = (const float*)d_in[7];
  float* out = (float*)d_out;

  // workspace layout (~42 MB):
  //   [0)      scale f32[512]
  //   [2048)   shift f32[512]
  //   [4096)   hpad  bf16[512]
  //   [8192)   h     bf16[16][1024][512]  (n-major, 16.78 MB)
  //   [+h]     duc   f32 [16][6][256][256] (25.17 MB)
  char* ws = (char*)d_ws;
  float*  scale = (float*)(ws + 0);
  float*  shift = (float*)(ws + 2048);
  __bf16* hpad  = (__bf16*)(ws + 4096);
  __bf16* h     = (__bf16*)(ws + 8192);
  float*  duc   = (float*)(ws + 8192 + (size_t)NBATCH * M1 * NSPAT * 2);

  prep_kernel<<<2, 256, 0, stream>>>(gamma, beta, mean, var, scale, shift, hpad);
  gemm1_bn_relu<<<dim3(8, 4, NBATCH), 256, 0, stream>>>(x, w1, scale, shift, h);
  gemm2_bias_duc<<<dim3(8, 3, NBATCH), 256, 0, stream>>>(h, w2, b2, hpad, duc);
  softmax_bilinear<<<4096, 256, 0, stream>>>(duc, out);
}
